// ScalarQuantizedXWLinear_46505905881434
// MI455X (gfx1250) — 8-bit matrix operand order repaired and checked by reading, compiled, not run on hardware
//
#include <hip/hip_runtime.h>
#include <stdint.h>

typedef int v8i __attribute__((ext_vector_type(8)));

__device__ __forceinline__ int quant1(float x, float inv) {
  float q = rintf(x * inv);
  q = fminf(127.0f, fmaxf(-127.0f, q));
  return (int)q;
}

__global__ void __launch_bounds__(256) quant_x_kernel(const float* __restrict__ x,
                                                      const float* __restrict__ scale_p,
                                                      int8_t* __restrict__ xq,
                                                      long long n4) {
  long long t = (long long)blockIdx.x * blockDim.x + threadIdx.x;
  if (t >= n4) return;
  const float s = fmaxf(scale_p[0], 1e-8f);
  const float inv = 1.0f / s;
  const float4 v = reinterpret_cast<const float4*>(x)[t];
  int q0 = quant1(v.x, inv) & 255;
  int q1 = quant1(v.y, inv) & 255;
  int q2 = quant1(v.z, inv) & 255;
  int q3 = quant1(v.w, inv) & 255;
  reinterpret_cast<int*>(xq)[t] = q0 | (q1 << 8) | (q2 << 16) | (q3 << 24);
}

__global__ void __launch_bounds__(256) quant_w_transpose_kernel(const float* __restrict__ w,
                                                                int8_t* __restrict__ wt,
                                                                int K, int N) {
  __shared__ int8_t tile[32][33];
  const int n0 = blockIdx.x * 32;
  const int k0 = blockIdx.y * 32;
  const int tx = threadIdx.x;
  const int ty = threadIdx.y;
#pragma unroll
  for (int i = ty; i < 32; i += 8) {
    float v = w[(size_t)(k0 + i) * N + (n0 + tx)];
    float q = fminf(127.0f, fmaxf(-127.0f, rintf(v)));
    tile[i][tx] = (int8_t)(int)q;
  }
  __syncthreads();
#pragma unroll
  for (int i = ty; i < 32; i += 8) {
    wt[(size_t)(n0 + i) * K + (k0 + tx)] = tile[tx][i];
  }
}

__device__ __forceinline__ v8i load_a_frag(const int8_t* __restrict__ rowp, int h) {
  v8i a;
#pragma unroll
  for (int v = 0; v < 8; ++v) {
    const int kb = ((v & 4) << 3) + ((v & 2) << 3) + ((v & 1) << 2) + (h << 3);
    a[v] = *reinterpret_cast<const int*>(rowp + kb);
  }
  return a;
}

__device__ __forceinline__ v8i load_b_frag(const int8_t* __restrict__ colp, int h) {
  v8i b;
#pragma unroll
  for (int v = 0; v < 8; ++v) {
    const int kb = ((v & 4) << 3) + ((v & 2) << 3) + ((v & 1) << 2) + (h << 3);
    b[v] = *reinterpret_cast<const int*>(colp + kb);
  }
  return b;
}

__global__ void __launch_bounds__(256) gemm_i8_wmma_kernel(const int8_t* __restrict__ xq,
                                                           const int8_t* __restrict__ wt,
                                                           const float* __restrict__ bias,
                                                           float* __restrict__ out,
                                                           int M, int N, int K) {
  const int lane = threadIdx.x & 31;
  const int wave = threadIdx.x >> 5;
  const int l15  = lane & 15;
  const int h    = lane >> 4;

  const int mBase = blockIdx.x * 128 + (wave & 1) * 64;
  const int nBase = blockIdx.y * 128 + (wave >> 1) * 32;

  const int8_t* aptr[4];
  const int8_t* bptr[2];
#pragma unroll
  for (int mt = 0; mt < 4; ++mt)
    aptr[mt] = xq + (size_t)(mBase + mt * 16 + l15) * K;
#pragma unroll
  for (int nt = 0; nt < 2; ++nt)
    bptr[nt] = wt + (size_t)(nBase + nt * 16 + l15) * K;

  v8i acc[4][2] = {};

  for (int kk = 0; kk < K; kk += 64) {
    __builtin_prefetch(aptr[0] + kk + 64, 0, 0);
    __builtin_prefetch(bptr[0] + kk + 64, 0, 0);

    v8i a[4], b[2];
#pragma unroll
    for (int mt = 0; mt < 4; ++mt) a[mt] = load_a_frag(aptr[mt] + kk, h);
#pragma unroll
    for (int nt = 0; nt < 2; ++nt) b[nt] = load_b_frag(bptr[nt] + kk, h);

#pragma unroll
    for (int mt = 0; mt < 4; ++mt)
#pragma unroll
      for (int nt = 0; nt < 2; ++nt)
        acc[mt][nt] = __builtin_amdgcn_wmma_i32_16x16x64_iu8(
             true, a[mt],  true, b[nt],
            acc[mt][nt],  false,  false);
  }

#pragma unroll
  for (int nt = 0; nt < 2; ++nt) {
    const int col = nBase + nt * 16 + l15;
    const float bv = bias[col];
#pragma unroll
    for (int mt = 0; mt < 4; ++mt) {
#pragma unroll
      for (int r = 0; r < 8; ++r) {
        const int row = mBase + mt * 16 + h * 8 + r;
        out[(size_t)row * N + col] = (float)acc[mt][nt][r] + bv;
      }
    }
  }
}

extern "C" void kernel_launch(void* const* d_in, const int* in_sizes, int n_in,
                              void* d_out, int out_size, void* d_ws, size_t ws_size,
                              hipStream_t stream) {
  const float* x    = (const float*)d_in[0];
  const float* w    = (const float*)d_in[1];
  const float* xs   = (const float*)d_in[2];
  const float* bias = (const float*)d_in[3];
  float* out = (float*)d_out;

  const int N = in_sizes[3];
  const int K = in_sizes[1] / N;
  const int M = in_sizes[0] / K;

  int8_t* xq = (int8_t*)d_ws;
  int8_t* wt = xq + (size_t)M * K;

  const long long n4 = ((long long)M * K) / 4;
  quant_x_kernel<<<(unsigned)((n4 + 255) / 256), 256, 0, stream>>>(x, xs, xq, n4);

  dim3 tb(32, 8);
  quant_w_transpose_kernel<<<dim3(N / 32, K / 32), tb, 0, stream>>>(w, wt, K, N);

  gemm_i8_wmma_kernel<<<dim3(M / 128, N / 128), 256, 0, stream>>>(xq, wt, bias, out, M, N, K);
}
